// ComplexAttention_3917010174114
// MI455X (gfx1250) — matrix operand order repaired and checked by reading, not compiled, not run on hardware
//
#include <hip/hip_runtime.h>
#include <hip/hip_bf16.h>
#include <stdint.h>

// Problem constants (from reference): B=4, T=256, H=32, D=1024
#define B_   4
#define T_   256
#define BT   1024          // B_*T_
#define H_   32
#define D_   1024
#define E_   2048          // 2*D_

typedef __bf16 bf16;
typedef __attribute__((ext_vector_type(16))) __bf16 v16bf;
typedef __attribute__((ext_vector_type(8)))  __bf16 v8bf;
typedef __attribute__((ext_vector_type(8)))  float  v8f;
typedef unsigned int u32x4 __attribute__((ext_vector_type(4)));
typedef int          i32x4 __attribute__((ext_vector_type(4)));
typedef int          i32x8 __attribute__((ext_vector_type(8)));

// ---------------------------------------------------------------------------
// TDM: 2D tile (rows x 32 bf16, global row stride in elements) -> LDS,
// with 4-DWORD pad every 16 DWORDs so LDS row stride is 40 bf16 (80B)
// => conflict-free strided fragment reads.
// ---------------------------------------------------------------------------
__device__ __forceinline__ void tdm_load_tile_2d(uint32_t lds_off,
                                                 const bf16* gptr,
                                                 uint32_t rows,
                                                 uint32_t row_elems,
                                                 uint64_t g_row_stride) {
  uint64_t ga = (uint64_t)(uintptr_t)gptr;
  u32x4 g0;
  g0[0] = 0x1u;                                   // count=1, user descriptor
  g0[1] = lds_off;                                // LDS byte address
  g0[2] = (uint32_t)ga;                           // global_addr[31:0]
  g0[3] = (uint32_t)((ga >> 32) & 0x01FFFFFFu)    // global_addr[56:32]
        | 0x80000000u;                            // type=2 ("image")
  uint32_t td0 = row_elems, td1 = rows;
  uint64_t st0 = g_row_stride;
  i32x8 g1;
  g1[0] = (int)((1u << 16)      // data_size = 2 bytes (bf16)
              | (1u << 20)      // pad_enable
              | (3u << 22)      // pad_interval: 16 DWORDs (one 32-elem row)
              | (3u << 25));    // pad_amount: 4 DWORDs (8 bf16)
  g1[1] = (int)((td0 & 0xFFFFu) << 16);                         // tensor_dim0[15:0]
  g1[2] = (int)((td0 >> 16) | ((td1 & 0xFFFFu) << 16));         // dim0 hi | dim1 lo
  g1[3] = (int)((td1 >> 16) | ((row_elems & 0xFFFFu) << 16));   // dim1 hi | tile_dim0
  g1[4] = (int)(rows & 0xFFFFu);                                // tile_dim1 (tile_dim2=0)
  g1[5] = (int)(uint32_t)st0;                                   // dim0_stride[31:0]
  g1[6] = (int)(uint32_t)((st0 >> 32) & 0xFFFFu);               // dim0_stride[47:32]
  g1[7] = 0;
  i32x4 gz = {0, 0, 0, 0};
#if __clang_major__ >= 23
  i32x8 gz8 = {0, 0, 0, 0, 0, 0, 0, 0};
  __builtin_amdgcn_tensor_load_to_lds(g0, g1, gz, gz, gz8, 0);
#else
  __builtin_amdgcn_tensor_load_to_lds(g0, g1, gz, gz, 0);
#endif
}

// ---------------------------------------------------------------------------
// bf16 WMMA GEMM: C(MxN, f32 accum) = A(MxK) @ Bt(NxK)^T [+ bias]
// Block 128x128, 8 waves (wave32), wave tile 32x64 (2x4 of 16x16), K-step 32.
// Double-buffered LDS: wave 0 issues the TDM pair for step i+1, then waits
// TENSORcnt<=2 (pair i done, pair i+1 in flight) -> DMA overlaps WMMA.
// ---------------------------------------------------------------------------
#define BLK_M 128
#define BLK_N 128
#define BLK_K 32
#define LDT   40   // padded LDS row stride in bf16 (32 data + 8 pad)

__global__ __launch_bounds__(256)
void gemm_bf16_wmma(const bf16* __restrict__ A,
                    const bf16* __restrict__ Bt,
                    const float* __restrict__ bias,
                    float* __restrict__ Cf,   // one of Cf/Cb is null
                    bf16*  __restrict__ Cb,
                    int M, int N, int K) {
  __shared__ __align__(16) bf16 ldsA[2][BLK_M * LDT];
  __shared__ __align__(16) bf16 ldsB[2][BLK_N * LDT];

  const int wave  = threadIdx.x >> 5;
  const int lane  = threadIdx.x & 31;
  const int wm    = wave >> 1;          // 0..3 along M
  const int wn    = wave & 1;           // 0..1 along N
  const int m0    = blockIdx.y * BLK_M;
  const int n0    = blockIdx.x * BLK_N;
  const int lhalf = lane >> 4;          // 0/1 lane half
  const int l16   = lane & 15;
  const int nk    = K / BLK_K;

  v8f acc[2][4];
#pragma unroll
  for (int i = 0; i < 2; ++i)
#pragma unroll
    for (int j = 0; j < 4; ++j)
      acc[i][j] = (v8f){0.f, 0.f, 0.f, 0.f, 0.f, 0.f, 0.f, 0.f};

  // Prologue: stage tiles for k-step 0 into buffer 0
  if (wave == 0) {
    tdm_load_tile_2d((uint32_t)(uintptr_t)&ldsA[0][0],
                     A + (size_t)m0 * K, BLK_M, BLK_K, (uint64_t)K);
    tdm_load_tile_2d((uint32_t)(uintptr_t)&ldsB[0][0],
                     Bt + (size_t)n0 * K, BLK_N, BLK_K, (uint64_t)K);
  }

  for (int i = 0; i < nk; ++i) {
    const int cur = i & 1;
    __syncthreads();                     // compute(i-1) done: buf[1-cur] reusable
    if (wave == 0) {                     // uniform branch -> one TDM pair per WG
      if (i + 1 < nk) {
        const int kk = (i + 1) * BLK_K;
        tdm_load_tile_2d((uint32_t)(uintptr_t)&ldsA[1 - cur][0],
                         A + (size_t)m0 * K + kk, BLK_M, BLK_K, (uint64_t)K);
        tdm_load_tile_2d((uint32_t)(uintptr_t)&ldsB[1 - cur][0],
                         Bt + (size_t)n0 * K + kk, BLK_N, BLK_K, (uint64_t)K);
        __builtin_amdgcn_s_wait_tensorcnt(2);   // pair(i) done, pair(i+1) in flight
      } else {
        __builtin_amdgcn_s_wait_tensorcnt(0);   // last tile: drain
      }
    }
    __syncthreads();                     // buf[cur] visible to all waves

    // A fragments: lane half 0 holds K{0..7,16..23}, half 1 holds K{8..15,24..31}
    v16bf af[2];
#pragma unroll
    for (int mi = 0; mi < 2; ++mi) {
      const bf16* ap = &ldsA[cur][(wm * 32 + mi * 16 + l16) * LDT + 8 * lhalf];
      v8bf a0 = *(const v8bf*)(ap);
      v8bf a1 = *(const v8bf*)(ap + 16);
      af[mi] = __builtin_shufflevector(a0, a1, 0, 1, 2, 3, 4, 5, 6, 7,
                                       8, 9, 10, 11, 12, 13, 14, 15);
    }
    // B fragments: lane half 0 holds K{0..7,16..23}, half 1 holds K{8..15,24..31}
#pragma unroll
    for (int ni = 0; ni < 4; ++ni) {
      const bf16* bp = &ldsB[cur][(wn * 64 + ni * 16 + l16) * LDT + 8 * lhalf];
      v8bf b0 = *(const v8bf*)(bp);
      v8bf b1 = *(const v8bf*)(bp + 16);
      v16bf bfrag = __builtin_shufflevector(b0, b1, 0, 1, 2, 3, 4, 5, 6, 7,
                                            8, 9, 10, 11, 12, 13, 14, 15);
#pragma unroll
      for (int mi = 0; mi < 2; ++mi)
        acc[mi][ni] = __builtin_amdgcn_wmma_f32_16x16x32_bf16(
            false, af[mi], false, bfrag, (short)0, acc[mi][ni], false, false);
    }
  }

  // Epilogue: C layout -> VGPR r, lane: M = r + 8*lhalf, N = l16
#pragma unroll
  for (int mi = 0; mi < 2; ++mi) {
#pragma unroll
    for (int ni = 0; ni < 4; ++ni) {
      const int n = n0 + wn * 64 + ni * 16 + l16;
      const float bval = bias ? bias[n] : 0.0f;
#pragma unroll
      for (int r = 0; r < 8; ++r) {
        const int m = m0 + wm * 32 + mi * 16 + r + lhalf * 8;
        const float v = acc[mi][ni][r] + bval;
        if (Cf) Cf[(size_t)m * N + n] = v;
        else    Cb[(size_t)m * N + n] = (bf16)v;
      }
    }
  }
}

// ---------------------------------------------------------------------------
// Prep kernels: build bf16 operands (concatenated cur, transposed weights)
// ---------------------------------------------------------------------------
__global__ void prep_curcat(const float* __restrict__ cr,
                            const float* __restrict__ ci,
                            bf16* __restrict__ out) {
  int idx = blockIdx.x * blockDim.x + threadIdx.x;
  if (idx >= BT * E_) return;
  int bt = idx / E_, e = idx - bt * E_;
  float v = (e < D_) ? cr[bt * D_ + e] : ci[bt * D_ + (e - D_)];
  out[idx] = (bf16)v;
}

// dst (R x C) = src^T where src is (C x R): dst[r*C + c] = src[c*R + r]
__global__ void prep_transpose_bf16(const float* __restrict__ src,
                                    bf16* __restrict__ dst, int R, int C) {
  int idx = blockIdx.x * blockDim.x + threadIdx.x;
  if (idx >= R * C) return;
  int r = idx / C, c = idx - r * C;
  dst[idx] = (bf16)src[(size_t)c * R + r];
}

__global__ void prep_cast_bf16(const float* __restrict__ src,
                               bf16* __restrict__ dst, int n) {
  int idx = blockIdx.x * blockDim.x + threadIdx.x;
  if (idx < n) dst[idx] = (bf16)src[idx];
}

// ---------------------------------------------------------------------------
// Streaming kernel: per (b,t) -> scores = scale*conf*(hist . qk), softmax(32),
// hbar = sum_h attn_h * hist_h   (bf16 out, feeds the ctx WMMA GEMM)
// ---------------------------------------------------------------------------
__global__ __launch_bounds__(256)
void attn_softmax_hbar(const float* __restrict__ hist_real,
                       const float* __restrict__ hist_imag,
                       const float* __restrict__ qk,
                       const float* __restrict__ conf,
                       bf16* __restrict__ hbar) {
  __shared__ float s_scores[H_];
  __shared__ float s_attn[H_];
  const int bt   = blockIdx.x;
  const int wave = threadIdx.x >> 5;
  const int lane = threadIdx.x & 31;
  const float* qv = qk + (size_t)bt * E_;
  const size_t hbase = (size_t)bt * H_ * D_;

  // Pass 1: 8 waves x 4 heads, dot over 2048 (real||imag), wave32 reduction
#pragma unroll
  for (int i = 0; i < 4; ++i) {
    const int h = wave * 4 + i;
    const float* hr = hist_real + hbase + (size_t)h * D_;
    const float* hi = hist_imag + hbase + (size_t)h * D_;
    float acc = 0.f;
    for (int e = lane; e < D_; e += 32) {
      __builtin_prefetch(hr + e + 512, 0, 0);
      __builtin_prefetch(hi + e + 512, 0, 0);
      acc = fmaf(hr[e], qv[e], acc);
      acc = fmaf(hi[e], qv[D_ + e], acc);
    }
#pragma unroll
    for (int off = 16; off > 0; off >>= 1) acc += __shfl_xor(acc, off, 32);
    if (lane == 0) s_scores[h] = acc;
  }
  __syncthreads();

  // Softmax over 32 heads in wave 0 (bk term is h-invariant: cancels)
  if (threadIdx.x < 32) {
    const float scale = 0.03125f;  // 1/sqrt(D), D=1024
    float s = s_scores[lane] * scale * conf[bt];
    float mx = s;
#pragma unroll
    for (int off = 16; off > 0; off >>= 1) mx = fmaxf(mx, __shfl_xor(mx, off, 32));
    float ex = __expf(s - mx);
    float sum = ex;
#pragma unroll
    for (int off = 16; off > 0; off >>= 1) sum += __shfl_xor(sum, off, 32);
    s_attn[lane] = ex / sum;
  }
  __syncthreads();

  // Pass 2: hbar[e] = sum_h attn[h]*hist[h][e]  (coalesced, L2-resident reread)
#pragma unroll
  for (int j = 0; j < E_ / 256; ++j) {
    const int e = threadIdx.x + j * 256;
    const float* src = (e < D_) ? (hist_real + hbase + e)
                                : (hist_imag + hbase + (e - D_));
    float acc = 0.f;
#pragma unroll 8
    for (int h = 0; h < H_; ++h) acc = fmaf(s_attn[h], src[(size_t)h * D_], acc);
    hbar[(size_t)bt * E_ + e] = (bf16)acc;
  }
}

// ---------------------------------------------------------------------------
// Final combine: complex64 out = (cur_real + 0.1*ctx_re, cur_imag + 0.1*ctx_im)
// ---------------------------------------------------------------------------
__global__ void combine_out(const float* __restrict__ cr,
                            const float* __restrict__ ci,
                            const float* __restrict__ ctx,
                            float* __restrict__ out) {
  int idx = blockIdx.x * blockDim.x + threadIdx.x;
  if (idx >= BT * D_) return;
  int bt = idx / D_, d = idx - bt * D_;
  out[2 * idx]     = cr[idx] + 0.1f * ctx[(size_t)bt * E_ + d];
  out[2 * idx + 1] = ci[idx] + 0.1f * ctx[(size_t)bt * E_ + D_ + d];
}

// ---------------------------------------------------------------------------
extern "C" void kernel_launch(void* const* d_in, const int* in_sizes, int n_in,
                              void* d_out, int out_size, void* d_ws, size_t ws_size,
                              hipStream_t stream) {
  const float* hist_real = (const float*)d_in[0];
  const float* hist_imag = (const float*)d_in[1];
  const float* cur_real  = (const float*)d_in[2];
  const float* cur_imag  = (const float*)d_in[3];
  const float* confid    = (const float*)d_in[4];
  const float* wq        = (const float*)d_in[5];
  const float* bq        = (const float*)d_in[6];
  const float* wk        = (const float*)d_in[7];
  // d_in[8] = bk: h-invariant score shift, cancels in softmax (and is zero)
  const float* wv        = (const float*)d_in[9];
  const float* bv        = (const float*)d_in[10];

  char* ws = (char*)d_ws;
  const size_t MB = 1024 * 1024;
  bf16*  curcat = (bf16*)(ws + 0 * MB);    // BT x E          (4 MB)
  bf16*  wqT    = (bf16*)(ws + 4 * MB);    // D x E  = wq^T   (4 MB)
  bf16*  wkB    = (bf16*)(ws + 8 * MB);    // E x D  = wk     (4 MB)
  bf16*  wvT    = (bf16*)(ws + 12 * MB);   // E x E  = wv^T   (8 MB)
  bf16*  q_bf   = (bf16*)(ws + 20 * MB);   // BT x D          (2 MB)
  float* qk_f   = (float*)(ws + 22 * MB);  // BT x E          (8 MB)
  bf16*  hbar   = (bf16*)(ws + 30 * MB);   // BT x E          (4 MB)
  float* ctx_f  = (float*)(ws + 34 * MB);  // BT x E          (8 MB)

  const int TB = 256;
  prep_curcat<<<(BT * E_ + TB - 1) / TB, TB, 0, stream>>>(cur_real, cur_imag, curcat);
  prep_transpose_bf16<<<(D_ * E_ + TB - 1) / TB, TB, 0, stream>>>(wq, wqT, D_, E_);
  prep_cast_bf16<<<(E_ * D_ + TB - 1) / TB, TB, 0, stream>>>(wk, wkB, E_ * D_);
  prep_transpose_bf16<<<(E_ * E_ + TB - 1) / TB, TB, 0, stream>>>(wv, wvT, E_, E_);

  // q = curcat @ wq + bq              (BT x D), bf16 out
  gemm_bf16_wmma<<<dim3(D_ / BLK_N, BT / BLK_M), TB, 0, stream>>>(
      curcat, wqT, bq, nullptr, q_bf, BT, D_, E_);
  // qk = q @ wk^T                     (BT x E), f32 out
  gemm_bf16_wmma<<<dim3(E_ / BLK_N, BT / BLK_M), TB, 0, stream>>>(
      q_bf, wkB, nullptr, qk_f, nullptr, BT, E_, D_);
  // scores/softmax/hbar: one WG per (b,t), streams the 268 MB of hist
  attn_softmax_hbar<<<BT, TB, 0, stream>>>(hist_real, hist_imag, qk_f, confid, hbar);
  // ctx = hbar @ wv + bv              (BT x E), f32 out
  gemm_bf16_wmma<<<dim3(E_ / BLK_N, BT / BLK_M), TB, 0, stream>>>(
      hbar, wvT, bv, ctx_f, nullptr, BT, E_, E_);
  // complex64 output, interleaved
  combine_out<<<(BT * D_ + TB - 1) / TB, TB, 0, stream>>>(cur_real, cur_imag, ctx_f,
                                                          (float*)d_out);
}
